// Decoder_75170517614716
// MI455X (gfx1250) — matrix operand order repaired and checked by reading, not compiled, not run on hardware
//
#include <hip/hip_runtime.h>

// ---------------------------------------------------------------------------
// Bahdanau-attention GRU decoder for MI455X (gfx1250, wave32, WMMA).
//
// Roofline: ~150 GFLOP of skinny (M=128) GEMMs; attention re-reads
// (proj_key bf16 16.8MB + encoder_hidden 67MB) stay L2-resident (192MB L2).
// All recurring GEMMs run on v_wmma_f32_16x16x32_bf16 with *pre-converted*
// bf16 operands so the K-loop is pure b128 loads + WMMA (no cvt VALU).
// B fragments for all 4 N-subtiles are preloaded before the 4 WMMAs so the
// scheduler can issue partial s_wait_loadcnt and overlap loads with WMMA.
// ---------------------------------------------------------------------------

typedef __attribute__((ext_vector_type(16))) __bf16     bf16x16;
typedef __attribute__((ext_vector_type(8)))  float      f32x8;
typedef __attribute__((ext_vector_type(8)))  unsigned   u32x8;

constexpr int B  = 128, S = 128, T = 128, E = 256, H = 512;
constexpr int H2 = 2 * H;          // 1024
constexpr int XK = E + H2;         // 1280  (GRU input  [embed, context])
constexpr int YK = E + H + H2;     // 1792  (preout in  [embed, h_new, context])
constexpr int G3 = 3 * H;          // 1536

constexpr size_t DS_OFF   = 0;
constexpr size_t HID_OFF  = (size_t)B * T * H;            // 8388608
constexpr size_t PRE_OFF  = HID_OFF + (size_t)B * H;      // 8454144
constexpr size_t CELL_OFF = PRE_OFF + (size_t)B * T * H;  // 16842752

// ---- f32 <-> bf16 helpers -------------------------------------------------
static __device__ __forceinline__ unsigned short f2bf(float f) {
  unsigned u = __float_as_uint(f);
  u += 0x7FFFu + ((u >> 16) & 1u);          // round to nearest even
  return (unsigned short)(u >> 16);
}
static __device__ __forceinline__ unsigned pack2(float lo, float hi) {
  return (unsigned)f2bf(lo) | ((unsigned)f2bf(hi) << 16);
}
static __device__ __forceinline__ float bflo(unsigned u) {
  return __uint_as_float(u << 16);
}
static __device__ __forceinline__ float bfhi(unsigned u) {
  return __uint_as_float(u & 0xFFFF0000u);
}
static __device__ __forceinline__ float sigmoidf(float x) {
  return 1.0f / (1.0f + __expf(-x));
}

// ---------------------------------------------------------------------------
// one-time f32 -> bf16 conversion (weights, encoder_final)
// ---------------------------------------------------------------------------
__global__ void __launch_bounds__(256)
conv_bf16(const float* __restrict__ src, unsigned short* __restrict__ dst, int n)
{
  for (int i = blockIdx.x * blockDim.x + threadIdx.x; i < n;
       i += gridDim.x * blockDim.x)
    dst[i] = f2bf(src[i]);
}

// ---------------------------------------------------------------------------
// STEADY-STATE GEMM: C[M,N] = act( A[M,K] * W[N,K]^T + bias )
// A, W pre-converted bf16; accum f32; C f32 (+ optional bf16 shadow Cb).
// One wave owns a 16x64 tile; A fragment reused across 4 v_wmma ops.
// Fragment layouts used here (wave32; contiguous K, same order for A and B):
//   A 16x32: lane l16 = row m0+l16; lanes 0-15 K k0..k0+15, lanes 16-31 K +16..+31
//   B 32x16: lane l16 = col n; lanes 0-15 K k0..k0+15, lanes 16-31 K +16..+31
//   C/D:     VGPR j -> row m0+j (lanes 0-15) / m0+8+j (lanes 16-31), col l16
// ---------------------------------------------------------------------------
__global__ void __launch_bounds__(256)
gemm_bf16(const unsigned short* __restrict__ A, int lda,
          const unsigned short* __restrict__ W, int ldw,
          float* __restrict__ C, long ldc,
          unsigned short* __restrict__ Cb, long ldcb,
          const float* __restrict__ bias,
          int M, int N, int K, int act)
{
  const int wavesPerBlock = blockDim.x >> 5;
  const int wave   = blockIdx.x * wavesPerBlock + (threadIdx.x >> 5);
  const int mTiles = M >> 4, nTiles = N >> 6;
  if (wave >= mTiles * nTiles) return;
  const int mt = wave % mTiles, nt = wave / mTiles;
  const int m0 = mt << 4, n0 = nt << 6;
  const int lane = threadIdx.x & 31, l16 = lane & 15, hi = lane >> 4;

  f32x8 acc[4] = {};
  const unsigned short* Arow = A + (size_t)(m0 + l16) * lda;
  const unsigned short* Wr0 = W + (size_t)(n0 +  0 + l16) * ldw + hi * 16;
  const unsigned short* Wr1 = W + (size_t)(n0 + 16 + l16) * ldw + hi * 16;
  const unsigned short* Wr2 = W + (size_t)(n0 + 32 + l16) * ldw + hi * 16;
  const unsigned short* Wr3 = W + (size_t)(n0 + 48 + l16) * ldw + hi * 16;

  for (int k0 = 0; k0 < K; k0 += 32) {
    // A fragment: two contiguous 8-bf16 (16B) chunks
    uint4 a0 = *(const uint4*)(Arow + k0 + hi * 16);
    uint4 a1 = *(const uint4*)(Arow + k0 + hi * 16 + 8);

    // preload ALL B fragments (8x b128) so waits can be partial
    uint4 bb[8];
    bb[0] = *(const uint4*)(Wr0 + k0);  bb[1] = *(const uint4*)(Wr0 + k0 + 8);
    bb[2] = *(const uint4*)(Wr1 + k0);  bb[3] = *(const uint4*)(Wr1 + k0 + 8);
    bb[4] = *(const uint4*)(Wr2 + k0);  bb[5] = *(const uint4*)(Wr2 + k0 + 8);
    bb[6] = *(const uint4*)(Wr3 + k0);  bb[7] = *(const uint4*)(Wr3 + k0 + 8);

    if (k0 + 32 < K) __builtin_prefetch(Arow + k0 + 32, 0, 3);  // near-scope prefetch

    u32x8 au;
    au[0] = a0.x; au[1] = a0.y; au[2] = a0.z; au[3] = a0.w;
    au[4] = a1.x; au[5] = a1.y; au[6] = a1.z; au[7] = a1.w;
    bf16x16 afrag = __builtin_bit_cast(bf16x16, au);

    #pragma unroll
    for (int nn = 0; nn < 4; ++nn) {
      u32x8 bu;
      bu[0] = bb[2*nn].x;   bu[1] = bb[2*nn].y;
      bu[2] = bb[2*nn].z;   bu[3] = bb[2*nn].w;
      bu[4] = bb[2*nn+1].x; bu[5] = bb[2*nn+1].y;
      bu[6] = bb[2*nn+1].z; bu[7] = bb[2*nn+1].w;
      bf16x16 bfrag = __builtin_bit_cast(bf16x16, bu);

      acc[nn] = __builtin_amdgcn_wmma_f32_16x16x32_bf16(
          false, afrag, false, bfrag, (short)0, acc[nn], false, false);
    }
  }

  #pragma unroll
  for (int nn = 0; nn < 4; ++nn) {
    const int col = n0 + nn * 16 + l16;
    const float bv = bias ? bias[col] : 0.0f;
    #pragma unroll
    for (int j = 0; j < 8; ++j) {
      const int row = m0 + j + hi * 8;
      float v = acc[nn][j] + bv;
      if (act == 1) v = tanhf(v);
      C[(size_t)row * ldc + col] = v;
      if (Cb) Cb[(size_t)row * ldcb + col] = f2bf(v);
    }
  }
}

// ---------------------------------------------------------------------------
// ONE-TIME GEMM (proj_key): f32 A/W converted in-register, bf16 output.
// proj_key[b,s,h] = sum_k enc[b,s,k] * W_key[h,k]   (16384 x 512 x 1024)
// ---------------------------------------------------------------------------
__global__ void __launch_bounds__(256)
gemm_f32_to_bf16(const float* __restrict__ A, int lda,
                 const float* __restrict__ W, int ldw,
                 unsigned short* __restrict__ C, long ldc,
                 int M, int N, int K)
{
  const int wavesPerBlock = blockDim.x >> 5;
  const int wave   = blockIdx.x * wavesPerBlock + (threadIdx.x >> 5);
  const int mTiles = M >> 4, nTiles = N >> 6;
  if (wave >= mTiles * nTiles) return;
  const int mt = wave % mTiles, nt = wave / mTiles;
  const int m0 = mt << 4, n0 = nt << 6;
  const int lane = threadIdx.x & 31, l16 = lane & 15, hi = lane >> 4;

  f32x8 acc[4] = {};
  const float* Arow = A + (size_t)(m0 + l16) * lda;

  for (int k0 = 0; k0 < K; k0 += 32) {
    const float4* pa0 = (const float4*)(Arow + k0 + hi * 16);
    const float4* pa1 = (const float4*)(Arow + k0 + hi * 16 + 8);
    float4 a0 = pa0[0], a1 = pa0[1], a2 = pa1[0], a3 = pa1[1];
    u32x8 au;
    au[0] = pack2(a0.x, a0.y); au[1] = pack2(a0.z, a0.w);
    au[2] = pack2(a1.x, a1.y); au[3] = pack2(a1.z, a1.w);
    au[4] = pack2(a2.x, a2.y); au[5] = pack2(a2.z, a2.w);
    au[6] = pack2(a3.x, a3.y); au[7] = pack2(a3.z, a3.w);
    bf16x16 afrag = __builtin_bit_cast(bf16x16, au);

    #pragma unroll
    for (int nn = 0; nn < 4; ++nn) {
      const float4* pw = (const float4*)(W + (size_t)(n0 + nn * 16 + l16) * ldw
                                           + k0 + hi * 16);
      float4 b0 = pw[0], b1 = pw[1], b2 = pw[2], b3 = pw[3];
      u32x8 bu;
      bu[0] = pack2(b0.x, b0.y); bu[1] = pack2(b0.z, b0.w);
      bu[2] = pack2(b1.x, b1.y); bu[3] = pack2(b1.z, b1.w);
      bu[4] = pack2(b2.x, b2.y); bu[5] = pack2(b2.z, b2.w);
      bu[6] = pack2(b3.x, b3.y); bu[7] = pack2(b3.z, b3.w);
      bf16x16 bfrag = __builtin_bit_cast(bf16x16, bu);

      acc[nn] = __builtin_amdgcn_wmma_f32_16x16x32_bf16(
          false, afrag, false, bfrag, (short)0, acc[nn], false, false);
    }
  }

  #pragma unroll
  for (int nn = 0; nn < 4; ++nn) {
    const int col = n0 + nn * 16 + l16;
    #pragma unroll
    for (int j = 0; j < 8; ++j) {
      const int row = m0 + j + hi * 8;
      C[(size_t)row * ldc + col] = f2bf(acc[nn][j]);
    }
  }
}

// ---------------------------------------------------------------------------
// scores[b,s] = sum_h tanh(q[b,h] + proj_key_bf16[b,s,h]) * v[h]  (masked)
// one wave per (b,s): lane l covers h = l*16..l*16+15 (2x b128 of bf16,
// fully coalesced 1KB per row); wave32 xor-reduce.
// ---------------------------------------------------------------------------
__global__ void __launch_bounds__(256)
attn_scores(const float* __restrict__ q, const unsigned short* __restrict__ pkb,
            const float* __restrict__ vE, const unsigned char* __restrict__ mask,
            float* __restrict__ scores)
{
  const int wave = blockIdx.x * (blockDim.x >> 5) + (threadIdx.x >> 5);
  const int b = wave / S, s = wave % S;
  const int lane = threadIdx.x & 31;
  const int h0 = lane * 16;

  const uint4* pr = (const uint4*)(pkb + ((size_t)b * S + s) * H + h0);
  uint4 p0 = pr[0], p1 = pr[1];
  const float4* qr = (const float4*)(q + (size_t)b * H + h0);
  float4 q0 = qr[0], q1 = qr[1], q2 = qr[2], q3 = qr[3];
  const float4* vr = (const float4*)(vE + h0);
  float4 v0 = vr[0], v1 = vr[1], v2 = vr[2], v3 = vr[3];

  float sum = 0.0f;
  sum += tanhf(q0.x + bflo(p0.x)) * v0.x + tanhf(q0.y + bfhi(p0.x)) * v0.y;
  sum += tanhf(q0.z + bflo(p0.y)) * v0.z + tanhf(q0.w + bfhi(p0.y)) * v0.w;
  sum += tanhf(q1.x + bflo(p0.z)) * v1.x + tanhf(q1.y + bfhi(p0.z)) * v1.y;
  sum += tanhf(q1.z + bflo(p0.w)) * v1.z + tanhf(q1.w + bfhi(p0.w)) * v1.w;
  sum += tanhf(q2.x + bflo(p1.x)) * v2.x + tanhf(q2.y + bfhi(p1.x)) * v2.y;
  sum += tanhf(q2.z + bflo(p1.y)) * v2.z + tanhf(q2.w + bfhi(p1.y)) * v2.w;
  sum += tanhf(q3.x + bflo(p1.z)) * v3.x + tanhf(q3.y + bfhi(p1.z)) * v3.y;
  sum += tanhf(q3.z + bflo(p1.w)) * v3.z + tanhf(q3.w + bfhi(p1.w)) * v3.w;

  #pragma unroll
  for (int off = 16; off >= 1; off >>= 1) sum += __shfl_xor(sum, off, 32);
  if (lane == 0)
    scores[(size_t)b * S + s] = mask[(size_t)b * S + s] ? sum : -1.0e9f;
}

// ---------------------------------------------------------------------------
// per-b softmax over S, then context = alphas @ encoder_hidden[b];
// context written in bf16 straight into the GRU input (xb) and preout (yb).
// ---------------------------------------------------------------------------
__global__ void __launch_bounds__(256)
softmax_context(const float* __restrict__ sc, const float* __restrict__ enc,
                unsigned short* __restrict__ xbb, unsigned short* __restrict__ ybb)
{
  __shared__ float alp[S];
  __shared__ float red[128];
  const int b = blockIdx.x, tid = threadIdx.x;
  if (tid < S) { float v = sc[(size_t)b * S + tid]; alp[tid] = v; red[tid] = v; }
  __syncthreads();
  for (int st = 64; st >= 1; st >>= 1) {
    if (tid < st) red[tid] = fmaxf(red[tid], red[tid + st]);
    __syncthreads();
  }
  const float mx = red[0];
  __syncthreads();
  if (tid < S) { float e = __expf(alp[tid] - mx); alp[tid] = e; red[tid] = e; }
  __syncthreads();
  for (int st = 64; st >= 1; st >>= 1) {
    if (tid < st) red[tid] += red[tid + st];
    __syncthreads();
  }
  const float inv = 1.0f / red[0];
  __syncthreads();
  if (tid < S) alp[tid] *= inv;
  __syncthreads();

  const float* eb = enc + (size_t)b * S * H2;
  #pragma unroll
  for (int i = 0; i < 4; ++i) {             // 2H = 1024 = 4 * 256
    const int d = tid + i * 256;
    float acc = 0.0f;
    for (int s2 = 0; s2 < S; ++s2)
      acc = fmaf(alp[s2], eb[(size_t)s2 * H2 + d], acc);
    const unsigned short cb = f2bf(acc);
    xbb[(size_t)b * XK + E + d]     = cb;
    ybb[(size_t)b * YK + E + H + d] = cb;
  }
}

// copy trg_embed[:, t, :] (f32) into the bf16 heads of xb and yb
__global__ void __launch_bounds__(256)
embed_copy(const float* __restrict__ te, int t,
           unsigned short* __restrict__ xbb, unsigned short* __restrict__ ybb)
{
  const int idx = blockIdx.x * blockDim.x + threadIdx.x;   // B*E
  const int b = idx / E, e = idx % E;
  const unsigned short v = f2bf(te[((size_t)b * T + t) * E + e]);
  xbb[(size_t)b * XK + e] = v;
  ybb[(size_t)b * YK + e] = v;
}

// GRU gates; writes h_next (f32 + bf16), decoder_states[:,t,:], yb's h slot
__global__ void __launch_bounds__(256)
gru_gate(const float* __restrict__ gi, const float* __restrict__ gh,
         const float* __restrict__ hc, float* __restrict__ hn,
         unsigned short* __restrict__ hbf, unsigned short* __restrict__ ybb,
         float* __restrict__ out, int t)
{
  const int idx = blockIdx.x * blockDim.x + threadIdx.x;   // B*H
  const int b = idx >> 9, n = idx & (H - 1);
  const size_t g = (size_t)b * G3;
  const float r  = sigmoidf(gi[g + n]         + gh[g + n]);
  const float z  = sigmoidf(gi[g + H + n]     + gh[g + H + n]);
  const float nn = tanhf   (gi[g + 2 * H + n] + r * gh[g + 2 * H + n]);
  const float hv = (1.0f - z) * nn + z * hc[idx];
  hn[idx]  = hv;
  hbf[idx] = f2bf(hv);
  out[DS_OFF + ((size_t)b * T + t) * H + n] = hv;
  ybb[(size_t)b * YK + E + n] = f2bf(hv);
}

__global__ void __launch_bounds__(256)
finalize(const float* __restrict__ hfin, float* __restrict__ out)
{
  const int i = blockIdx.x * blockDim.x + threadIdx.x;     // B*H
  out[HID_OFF + i]  = hfin[i];
  out[CELL_OFF + i] = 0.0f;
}

// ---------------------------------------------------------------------------
extern "C" void kernel_launch(void* const* d_in, const int* in_sizes, int n_in,
                              void* d_out, int out_size, void* d_ws, size_t ws_size,
                              hipStream_t stream)
{
  const float* trg_embed  = (const float*)d_in[0];
  const float* enc_hidden = (const float*)d_in[1];
  const float* enc_final  = (const float*)d_in[2];   // (1,B,2H) -> (B,2H)
  const unsigned char* src_mask = (const unsigned char*)d_in[3];
  /* d_in[4] trg_mask: unused by the reference math */
  const float* W_key    = (const float*)d_in[5];
  const float* W_query  = (const float*)d_in[6];
  const float* v_energy = (const float*)d_in[7];
  const float* W_ih     = (const float*)d_in[8];
  const float* W_hh     = (const float*)d_in[9];
  const float* b_ih     = (const float*)d_in[10];
  const float* b_hh     = (const float*)d_in[11];
  const float* W_bridge = (const float*)d_in[12];
  const float* b_bridge = (const float*)d_in[13];
  const float* W_preout = (const float*)d_in[14];

  float* out = (float*)d_out;

  // ---- scratch layout (~29 MB) ----
  char* base = (char*)d_ws;
  size_t off = 0;
  auto salloc = [&](size_t bytes) {
    void* p = base + off;
    off = (off + bytes + 255) & ~size_t(255);
    return p;
  };
  unsigned short* pkb   = (unsigned short*)salloc((size_t)B * S * H * 2);  // 16.8MB
  unsigned short* wq_b  = (unsigned short*)salloc((size_t)H * H * 2);
  unsigned short* wih_b = (unsigned short*)salloc((size_t)G3 * XK * 2);
  unsigned short* whh_b = (unsigned short*)salloc((size_t)G3 * H * 2);
  unsigned short* wbr_b = (unsigned short*)salloc((size_t)H * H2 * 2);
  unsigned short* wpre_b= (unsigned short*)salloc((size_t)H * YK * 2);
  unsigned short* efb   = (unsigned short*)salloc((size_t)B * H2 * 2);
  unsigned short* hbf   = (unsigned short*)salloc((size_t)B * H * 2);
  unsigned short* xbb   = (unsigned short*)salloc((size_t)B * XK * 2);
  unsigned short* ybb   = (unsigned short*)salloc((size_t)B * YK * 2);
  float* hA = (float*)salloc((size_t)B * H * 4);
  float* hB = (float*)salloc((size_t)B * H * 4);
  float* qb = (float*)salloc((size_t)B * H * 4);
  float* sc = (float*)salloc((size_t)B * S * 4);
  float* gi = (float*)salloc((size_t)B * G3 * 4);
  float* gh = (float*)salloc((size_t)B * G3 * 4);

  auto conv = [&](const float* s, unsigned short* d, int n) {
    conv_bf16<<<(n + 255) / 256 < 1024 ? (n + 255) / 256 : 1024, 256, 0, stream>>>(s, d, n);
  };
  auto gemm = [&](const unsigned short* A, int lda, const unsigned short* W, int ldw,
                  float* C, long ldc, unsigned short* Cb, long ldcb,
                  const float* bias, int M, int N, int K, int act) {
    const int tiles  = (M / 16) * (N / 64);
    const int blocks = (tiles + 7) / 8;        // 8 waves (tiles) per 256-thread block
    gemm_bf16<<<blocks, 256, 0, stream>>>(A, lda, W, ldw, C, ldc, Cb, ldcb,
                                          bias, M, N, K, act);
  };

  // ---- one-time: weight/operand conversion + proj_key + bridge ----
  conv(W_query,  wq_b,  H * H);
  conv(W_ih,     wih_b, G3 * XK);
  conv(W_hh,     whh_b, G3 * H);
  conv(W_bridge, wbr_b, H * H2);
  conv(W_preout, wpre_b, H * YK);
  conv(enc_final, efb,  B * H2);

  {   // proj_key (bf16 out) = encoder_hidden @ W_key^T : 16384 x 512 x 1024
    const int tiles = (B * S / 16) * (H / 64);
    gemm_f32_to_bf16<<<(tiles + 7) / 8, 256, 0, stream>>>(
        enc_hidden, H2, W_key, H2, pkb, H, B * S, H, H2);
  }
  // h0 = tanh(encoder_final @ W_bridge^T + b_bridge)  -> hA (f32) + hbf (bf16)
  gemm(efb, H2, wbr_b, H2, hA, H, hbf, H, b_bridge, B, H, H2, 1);

  for (int t = 0; t < T; ++t) {
    float* hc = (t & 1) ? hB : hA;
    float* hn = (t & 1) ? hA : hB;

    // q = h @ W_query^T
    gemm(hbf, H, wq_b, H, qb, H, nullptr, 0, nullptr, B, H, H, 0);
    // attention energies + softmax + context (bf16 into xb/yb)
    attn_scores<<<(B * S) / 8, 256, 0, stream>>>(qb, pkb, v_energy, src_mask, sc);
    softmax_context<<<B, 256, 0, stream>>>(sc, enc_hidden, xbb, ybb);
    embed_copy<<<(B * E) / 256, 256, 0, stream>>>(trg_embed, t, xbb, ybb);
    // GRU: gi = [embed,ctx] @ W_ih^T + b_ih ; gh = h @ W_hh^T + b_hh
    gemm(xbb, XK, wih_b, XK, gi, G3, nullptr, 0, b_ih, B, G3, XK, 0);
    gemm(hbf, H, whh_b, H, gh, G3, nullptr, 0, b_hh, B, G3, H, 0);
    // gates: updates hn (f32), hbf (bf16, consumed by next step), yb h-slot
    gru_gate<<<(B * H) / 256, 256, 0, stream>>>(gi, gh, hc, hn, hbf, ybb, out, t);
    // pre_output[:,t,:] = [embed, h_new, ctx] @ W_preout^T  (strided C)
    gemm(ybb, YK, wpre_b, YK, out + PRE_OFF + (size_t)t * H, (long)T * H,
         nullptr, 0, nullptr, B, H, YK, 0);
  }

  // after t=127 (odd), h_next landed in hA
  finalize<<<(B * H) / 256, 256, 0, stream>>>(hA, out);
}
